// SparseQwenAttention_71253507441024
// MI455X (gfx1250) — matrix operand order repaired and checked by reading, not compiled, not run on hardware
//
#include <hip/hip_runtime.h>
#include <math.h>

// ---------------------------------------------------------------------------
// SparseQwenAttention on MI455X (gfx1250, wave32, WMMA).
// Heavy math (8x 2048^3 mask GEMMs, 4x 4096x2048x2048 projections, flash
// attention) runs on v_wmma_f32_16x16x32_bf16 with f32 accumulation.
// GEMM: LDS double-buffered pipeline, 32x64 per-wave tile so every LDS
// fragment feeds >=2 WMMAs (8 WMMAs per K-slice per wave).
// ---------------------------------------------------------------------------

#define H_DIM   2048
#define BATCH   2
#define SEQ     2048
#define NHEADS  16
#define HDIM    128
#define BS      (BATCH * SEQ)          // 4096
#define LPAD    40                     // padded LDS row stride (elements)

typedef __attribute__((ext_vector_type(16))) __bf16 v16bf;
typedef __attribute__((ext_vector_type(8)))  float  v8f;

union FragBF {
    v16bf v;
    uint4 u[2];
};

__device__ __forceinline__ unsigned short f2bf(float f) {
    union { float f; unsigned u; } x; x.f = f;
    unsigned r = x.u + 0x7FFFu + ((x.u >> 16) & 1u);   // round-to-nearest-even
    return (unsigned short)(r >> 16);
}

__device__ __forceinline__ v8f wmma_bf16(const FragBF& a, const FragBF& b, v8f c) {
    return __builtin_amdgcn_wmma_f32_16x16x32_bf16(
        /*neg_a=*/false, a.v, /*neg_b=*/false, b.v,
        /*c_mod=*/(short)0, c, /*reuse_a=*/false, /*reuse_b=*/false);
}

// ---------------------------------------------------------------------------
// Generic NT GEMM: C[M,N] = A[MxK] * B[NxK]^T (+bias[n]).  A,B bf16, C f32.
// Block = 256 thr = 8 waves; block tile 256(M) x 64(N); K step 32.
// Wave tile 32x64: 2 A-frags x 4 B-frags -> 8 WMMAs per slice.
// Double-buffered LDS staging: global loads for slice i+1 issue before the
// WMMAs of slice i; only the ds_stores wait on them.
// ---------------------------------------------------------------------------
__global__ void __launch_bounds__(256)
k_gemm_nt(const unsigned short* __restrict__ A,
          const unsigned short* __restrict__ B,
          float* __restrict__ C,
          const float* __restrict__ bias,
          int M, int N, int K)
{
    __shared__ __align__(16) unsigned short lA[2][256 * LPAD];
    __shared__ __align__(16) unsigned short lB[2][64 * LPAD];

    const int tid  = threadIdx.x;
    const int lane = tid & 31;
    const int wave = tid >> 5;
    const int half = lane >> 4;
    const int l16  = lane & 15;

    const long mBase = (long)blockIdx.y * 256;
    const long nBase = (long)blockIdx.x * 64;

    // Tile copy assignments: 16B chunks. A tile = 256 rows x 4 chunks = 1024
    // chunks (4 per thread); B tile = 64 rows x 4 chunks = 256 (1 per thread).
    const int ar  = tid >> 2;            // base row 0..63 (strided by 64)
    const int akc = (tid & 3) * 8;       // k-chunk offset in elements

    const unsigned short* gA0 = A + (mBase + ar +   0) * (long)K + akc;
    const unsigned short* gA1 = A + (mBase + ar +  64) * (long)K + akc;
    const unsigned short* gA2 = A + (mBase + ar + 128) * (long)K + akc;
    const unsigned short* gA3 = A + (mBase + ar + 192) * (long)K + akc;
    const unsigned short* gB  = B + (nBase + ar) * (long)K + akc;

    // Preload slice 0 into buffer 0.
    {
        uint4 r0 = *reinterpret_cast<const uint4*>(gA0);
        uint4 r1 = *reinterpret_cast<const uint4*>(gA1);
        uint4 r2 = *reinterpret_cast<const uint4*>(gA2);
        uint4 r3 = *reinterpret_cast<const uint4*>(gA3);
        uint4 rb = *reinterpret_cast<const uint4*>(gB);
        *reinterpret_cast<uint4*>(&lA[0][(ar +   0) * LPAD + akc]) = r0;
        *reinterpret_cast<uint4*>(&lA[0][(ar +  64) * LPAD + akc]) = r1;
        *reinterpret_cast<uint4*>(&lA[0][(ar + 128) * LPAD + akc]) = r2;
        *reinterpret_cast<uint4*>(&lA[0][(ar + 192) * LPAD + akc]) = r3;
        *reinterpret_cast<uint4*>(&lB[0][ar * LPAD + akc]) = rb;
    }
    __syncthreads();

    v8f c00 = {}, c01 = {}, c02 = {}, c03 = {};
    v8f c10 = {}, c11 = {}, c12 = {}, c13 = {};
    int buf = 0;

    for (int k0 = 0; k0 < K; k0 += 32) {
        const bool more = (k0 + 32) < K;

        // Issue global loads for the NEXT slice (in flight during compute).
        uint4 n0, n1, n2, n3, nb;
        if (more) {
            n0 = *reinterpret_cast<const uint4*>(gA0 + k0 + 32);
            n1 = *reinterpret_cast<const uint4*>(gA1 + k0 + 32);
            n2 = *reinterpret_cast<const uint4*>(gA2 + k0 + 32);
            n3 = *reinterpret_cast<const uint4*>(gA3 + k0 + 32);
            nb = *reinterpret_cast<const uint4*>(gB  + k0 + 32);
        }
        if (k0 + 256 < K) {                       // far prefetch (gfx1250)
            __builtin_prefetch(gA0 + k0 + 256, 0, 1);
            __builtin_prefetch(gB  + k0 + 256, 0, 1);
        }

        // Compute on current buffer (LDS fragment reads, bank-conflict free).
        const unsigned short* sa0 = &lA[buf][(wave * 32 + l16) * LPAD];
        const unsigned short* sa1 = sa0 + 16 * LPAD;
        FragBF a0, a1, b0, b1, b2, b3;
        a0.u[0] = *reinterpret_cast<const uint4*>(sa0 + half * 8);
        a0.u[1] = *reinterpret_cast<const uint4*>(sa0 + 16 + half * 8);
        a1.u[0] = *reinterpret_cast<const uint4*>(sa1 + half * 8);
        a1.u[1] = *reinterpret_cast<const uint4*>(sa1 + 16 + half * 8);
        const unsigned short* sb0 = &lB[buf][( 0 + l16) * LPAD + half * 8];
        const unsigned short* sb1 = &lB[buf][(16 + l16) * LPAD + half * 8];
        const unsigned short* sb2 = &lB[buf][(32 + l16) * LPAD + half * 8];
        const unsigned short* sb3 = &lB[buf][(48 + l16) * LPAD + half * 8];
        b0.u[0] = *reinterpret_cast<const uint4*>(sb0);
        b0.u[1] = *reinterpret_cast<const uint4*>(sb0 + 16);
        b1.u[0] = *reinterpret_cast<const uint4*>(sb1);
        b1.u[1] = *reinterpret_cast<const uint4*>(sb1 + 16);
        b2.u[0] = *reinterpret_cast<const uint4*>(sb2);
        b2.u[1] = *reinterpret_cast<const uint4*>(sb2 + 16);
        b3.u[0] = *reinterpret_cast<const uint4*>(sb3);
        b3.u[1] = *reinterpret_cast<const uint4*>(sb3 + 16);

        // Each B fragment feeds two WMMAs back-to-back.
        c00 = wmma_bf16(a0, b0, c00);
        c10 = wmma_bf16(a1, b0, c10);
        c01 = wmma_bf16(a0, b1, c01);
        c11 = wmma_bf16(a1, b1, c11);
        c02 = wmma_bf16(a0, b2, c02);
        c12 = wmma_bf16(a1, b2, c12);
        c03 = wmma_bf16(a0, b3, c03);
        c13 = wmma_bf16(a1, b3, c13);

        // Store next slice into the other buffer (waits on global loads here,
        // after the WMMAs were issued).
        if (more) {
            buf ^= 1;
            *reinterpret_cast<uint4*>(&lA[buf][(ar +   0) * LPAD + akc]) = n0;
            *reinterpret_cast<uint4*>(&lA[buf][(ar +  64) * LPAD + akc]) = n1;
            *reinterpret_cast<uint4*>(&lA[buf][(ar + 128) * LPAD + akc]) = n2;
            *reinterpret_cast<uint4*>(&lA[buf][(ar + 192) * LPAD + akc]) = n3;
            *reinterpret_cast<uint4*>(&lB[buf][ar * LPAD + akc]) = nb;
        }
        __syncthreads();
    }

    const float bv0 = bias ? bias[nBase +  0 + l16] : 0.0f;
    const float bv1 = bias ? bias[nBase + 16 + l16] : 0.0f;
    const float bv2 = bias ? bias[nBase + 32 + l16] : 0.0f;
    const float bv3 = bias ? bias[nBase + 48 + l16] : 0.0f;

    const long mrow0 = mBase + wave * 32;
#pragma unroll
    for (int r = 0; r < 8; ++r) {
        long mr = mrow0 + r + 8 * half;          // C layout: row = r + 8*half
        float* cp = C + mr * (long)N + nBase + l16;
        cp[0]  = c00[r] + bv0;
        cp[16] = c01[r] + bv1;
        cp[32] = c02[r] + bv2;
        cp[48] = c03[r] + bv3;
        float* cq = cp + 16 * (long)N;           // second 16-row subtile
        cq[0]  = c10[r] + bv0;
        cq[16] = c11[r] + bv1;
        cq[32] = c12[r] + bv2;
        cq[48] = c13[r] + bv3;
    }
}

// ---------------------------------------------------------------------------
// Flash attention. Q,K bf16 [B,NH,S,D]; Vt bf16 [B,NH,D,S]; out bf16 [B,S,NH*D].
// Block = 128 thr = 4 waves; block covers 64 query rows; wave owns 16.
// Fragment loads are batched (one clause + one wait per WMMA group).
// ---------------------------------------------------------------------------
__global__ void __launch_bounds__(128)
k_flash(const unsigned short* __restrict__ Q,
        const unsigned short* __restrict__ Kc,
        const unsigned short* __restrict__ Vt,
        unsigned short* __restrict__ O,
        int S)
{
    __shared__ __align__(16) unsigned short pbuf[4][16 * 32]; // per-wave P staging

    const int lane = threadIdx.x & 31;
    const int wave = threadIdx.x >> 5;
    const int half = lane >> 4;
    const int l16  = lane & 15;

    const int bh = blockIdx.y;                    // b*NHEADS + h
    const int b  = bh >> 4;
    const int h  = bh & 15;
    const int q0 = blockIdx.x * 64;
    const int qw = q0 + wave * 16;

    const long base = (long)bh * S * HDIM;        // same extent for Q/K/Vt
    const unsigned short* qrow = Q + base + (long)(qw + l16) * HDIM;

    FragBF qf[4];
#pragma unroll
    for (int c = 0; c < 4; ++c) {
        qf[c].u[0] = *reinterpret_cast<const uint4*>(qrow + c * 32 + half * 8);
        qf[c].u[1] = *reinterpret_cast<const uint4*>(qrow + c * 32 + 16 + half * 8);
    }

    v8f o[8];
    float mrow[8], lrow[8];
#pragma unroll
    for (int d = 0; d < 8; ++d) o[d] = (v8f){};
#pragma unroll
    for (int r = 0; r < 8; ++r) { mrow[r] = -3.0e38f; lrow[r] = 0.0f; }

    const float scale = 0.08838834764831845f;     // 1/sqrt(128)
    const int kend = q0 + 64;                     // causal bound (block-uniform)

    for (int kt = 0; kt < kend; kt += 32) {
        // ---- batched K-fragment loads, then S = Q * K^T ----
        const unsigned short* kr0 = Kc + base + (long)(kt + l16) * HDIM;
        const unsigned short* kr1 = Kc + base + (long)(kt + 16 + l16) * HDIM;
        FragBF kf0[4], kf1[4];
#pragma unroll
        for (int c = 0; c < 4; ++c) {
            kf0[c].u[0] = *reinterpret_cast<const uint4*>(kr0 + c * 32 + half * 8);
            kf0[c].u[1] = *reinterpret_cast<const uint4*>(kr0 + c * 32 + 16 + half * 8);
            kf1[c].u[0] = *reinterpret_cast<const uint4*>(kr1 + c * 32 + half * 8);
            kf1[c].u[1] = *reinterpret_cast<const uint4*>(kr1 + c * 32 + 16 + half * 8);
        }
        v8f s0 = {}, s1 = {};
#pragma unroll
        for (int c = 0; c < 4; ++c) {
            s0 = wmma_bf16(qf[c], kf0[c], s0);
            s1 = wmma_bf16(qf[c], kf1[c], s1);
        }

        // ---- causal mask, scale, online softmax ----
        float mnew[8];
#pragma unroll
        for (int r = 0; r < 8; ++r) {
            int qg  = qw + r + 8 * half;
            int kg0 = kt + l16;
            int kg1 = kt + 16 + l16;
            float v0 = (kg0 <= qg) ? s0[r] * scale : -3.0e38f;
            float v1 = (kg1 <= qg) ? s1[r] * scale : -3.0e38f;
            s0[r] = v0; s1[r] = v1;
            float rm = fmaxf(v0, v1);
            rm = fmaxf(rm, __shfl_xor(rm, 1));
            rm = fmaxf(rm, __shfl_xor(rm, 2));
            rm = fmaxf(rm, __shfl_xor(rm, 4));
            rm = fmaxf(rm, __shfl_xor(rm, 8));
            mnew[r] = fmaxf(mrow[r], rm);
        }
#pragma unroll
        for (int r = 0; r < 8; ++r) {
            float p0 = __expf(s0[r] - mnew[r]);
            float p1 = __expf(s1[r] - mnew[r]);
            s0[r] = p0; s1[r] = p1;
            float rs = p0 + p1;
            rs += __shfl_xor(rs, 1);
            rs += __shfl_xor(rs, 2);
            rs += __shfl_xor(rs, 4);
            rs += __shfl_xor(rs, 8);
            float alpha = __expf(mrow[r] - mnew[r]);
            lrow[r] = lrow[r] * alpha + rs;
            mrow[r] = mnew[r];
#pragma unroll
            for (int d = 0; d < 8; ++d) o[d][r] *= alpha;
        }

        // ---- stage P through LDS: C-layout -> A-fragment layout ----
        unsigned short* pw = pbuf[wave];
#pragma unroll
        for (int r = 0; r < 8; ++r) {
            int row = r + 8 * half;
            pw[row * 32 + l16]      = f2bf(s0[r]);
            pw[row * 32 + 16 + l16] = f2bf(s1[r]);
        }
        FragBF pf;
        pf.u[0] = *reinterpret_cast<const uint4*>(pw + l16 * 32 + half * 8);
        pf.u[1] = *reinterpret_cast<const uint4*>(pw + l16 * 32 + 16 + half * 8);

        // ---- batched V-fragment loads, then O += P * V ----
        FragBF vf[8];
#pragma unroll
        for (int d = 0; d < 8; ++d) {
            const unsigned short* vr = Vt + base + (long)(d * 16 + l16) * S + kt;
            vf[d].u[0] = *reinterpret_cast<const uint4*>(vr + half * 8);
            vf[d].u[1] = *reinterpret_cast<const uint4*>(vr + 16 + half * 8);
        }
#pragma unroll
        for (int d = 0; d < 8; ++d) o[d] = wmma_bf16(pf, vf[d], o[d]);
    }

    // ---- normalize + store [B,S,NH*D] bf16 ----
#pragma unroll
    for (int r = 0; r < 8; ++r) {
        float inv = 1.0f / lrow[r];
        int srow = qw + r + 8 * half;
        unsigned short* op = O + ((long)(b * S + srow) * (NHEADS * HDIM)) + h * HDIM + l16;
#pragma unroll
        for (int d = 0; d < 8; ++d) op[d * 16] = f2bf(o[d][r] * inv);
    }
}

// ---------------------------------------------------------------------------
// Small elementwise / reduction kernels
// ---------------------------------------------------------------------------
__global__ void k_colsum(const float* __restrict__ W, float* __restrict__ cs, int R, int C) {
    int j = blockIdx.x * blockDim.x + threadIdx.x;
    if (j >= C) return;
    float s = 0.0f;
    for (int i = 0; i < R; ++i) s += fabsf(W[(long)i * C + j]);
    cs[j] = s;
}

__global__ void __launch_bounds__(256) k_rowsum(const float* __restrict__ W, float* __restrict__ rs, int C) {
    __shared__ float red[256];
    const float* wr = W + (long)blockIdx.x * C;
    float s = 0.0f;
    for (int j = threadIdx.x; j < C; j += 256) s += fabsf(wr[j]);
    red[threadIdx.x] = s;
    __syncthreads();
    for (int off = 128; off > 0; off >>= 1) {
        if (threadIdx.x < off) red[threadIdx.x] += red[threadIdx.x + off];
        __syncthreads();
    }
    if (threadIdx.x == 0) rs[blockIdx.x] = red[0];
}

__global__ void k_metric(const float* __restrict__ W, const float* __restrict__ cs,
                         const float* __restrict__ rs, const float* __restrict__ sr,
                         unsigned short* __restrict__ out, int C, long n) {
    long idx = (long)blockIdx.x * blockDim.x + threadIdx.x;
    if (idx >= n) return;
    int i = (int)(idx / C), j = (int)(idx % C);
    float a = fabsf(W[idx]);
    float m = (a / cs[j] + a / rs[i]) * sqrtf(sr[j]);   // RIA metric, a=1.0
    out[idx] = f2bf(m);
}

__global__ void k_hard(const float* __restrict__ Wm, unsigned short* __restrict__ hard, long nblk) {
    long t = (long)blockIdx.x * blockDim.x + threadIdx.x;
    if (t >= nblk) return;
    const float* p = Wm + t * 4;
    float v0 = p[0], v1 = p[1], v2 = p[2], v3 = p[3];
    int i1 = 0; float best = v0;
    if (v1 > best) { best = v1; i1 = 1; }
    if (v2 > best) { best = v2; i1 = 2; }
    if (v3 > best) { best = v3; i1 = 3; }
    int i2 = -1; float b2 = -3.0e38f;
#pragma unroll
    for (int i = 0; i < 4; ++i) {
        if (i == i1) continue;
        float v = (i == 0) ? v0 : (i == 1) ? v1 : (i == 2) ? v2 : v3;
        if (v > b2) { b2 = v; i2 = i; }
    }
    unsigned short* o = hard + t * 4;
#pragma unroll
    for (int i = 0; i < 4; ++i) o[i] = (i == i1 || i == i2) ? (unsigned short)0x3F80 : (unsigned short)0;
}

__global__ void k_applymask(const float* __restrict__ W, const float* __restrict__ maskP,
                            unsigned short* __restrict__ out, long n) {
    long idx = (long)blockIdx.x * blockDim.x + threadIdx.x;
    if (idx >= n) return;
    out[idx] = f2bf(W[idx] * maskP[idx]);
}

__global__ void k_f32_to_bf16(const float* __restrict__ in, unsigned short* __restrict__ out, long n) {
    long idx = (long)blockIdx.x * blockDim.x + threadIdx.x;
    if (idx >= n) return;
    out[idx] = f2bf(in[idx]);
}

__global__ void k_transpose_bf16(const float* __restrict__ in, unsigned short* __restrict__ out, int H) {
    long idx = (long)blockIdx.x * blockDim.x + threadIdx.x;
    if (idx >= (long)H * H) return;
    int n = (int)(idx / H), k = (int)(idx % H);
    out[idx] = f2bf(in[(long)k * H + n]);
}

// proj f32 [B,S,NH,D] -> RoPE -> bf16 [B,NH,S,D]
__global__ void k_rope(const float* __restrict__ x, const long long* __restrict__ pos,
                       unsigned short* __restrict__ out, int S) {
    long idx = (long)blockIdx.x * blockDim.x + threadIdx.x;
    long total = (long)BATCH * S * NHEADS * HDIM;
    if (idx >= total) return;
    int d = (int)(idx & 127);
    int h = (int)((idx >> 7) & 15);
    long bs = idx >> 11;
    int s = (int)(bs % S);
    int b = (int)(bs / S);
    float p = (float)pos[bs];
    int j = d & 63;
    float inv = __expf(-(2.0f * j / 128.0f) * 9.210340371976184f);  // 10000^(-2j/128)
    float ang = p * inv;
    float sn, cn;
    __sincosf(ang, &sn, &cn);
    float xv = x[idx];
    float rot = (d < 64) ? -x[idx + 64] : x[idx - 64];
    float y = xv * cn + rot * sn;
    out[(((long)b * NHEADS + h) * S + s) * HDIM + d] = f2bf(y);
}

// proj f32 [B,S,NH,D] -> bf16 [B,NH,D,S] (transposed values for P@V fragments)
__global__ void k_vt(const float* __restrict__ x, unsigned short* __restrict__ out, int S) {
    long idx = (long)blockIdx.x * blockDim.x + threadIdx.x;
    long total = (long)BATCH * S * NHEADS * HDIM;
    if (idx >= total) return;
    int d = (int)(idx & 127);
    int h = (int)((idx >> 7) & 15);
    long bs = idx >> 11;
    int s = (int)(bs % S);
    int b = (int)(bs / S);
    out[(((long)b * NHEADS + h) * HDIM + d) * S + s] = f2bf(x[idx]);
}

// ---------------------------------------------------------------------------
// Host-side orchestration
// ---------------------------------------------------------------------------
extern "C" void kernel_launch(void* const* d_in, const int* in_sizes, int n_in,
                              void* d_out, int out_size, void* d_ws, size_t ws_size,
                              hipStream_t stream) {
    const int   H  = H_DIM;
    const long  HH = (long)H * H;
    const long  BSH = (long)BS * H;

    const float* hs   = (const float*)d_in[0];
    const float* Wq   = (const float*)d_in[1];
    const float* bq   = (const float*)d_in[2];
    const float* Wk   = (const float*)d_in[3];
    const float* bk   = (const float*)d_in[4];
    const float* Wv   = (const float*)d_in[5];
    const float* bv   = (const float*)d_in[6];
    const float* Wo   = (const float*)d_in[7];
    const float* sq   = (const float*)d_in[8];
    const float* sk   = (const float*)d_in[9];
    const float* sv   = (const float*)d_in[10];
    const float* so   = (const float*)d_in[11];
    const float* pqkv = (const float*)d_in[12];
    const float* po   = (const float*)d_in[13];
    const long long* pos = (const long long*)d_in[14];

    char* w = (char*)d_ws;
    size_t off = 0;
    auto take = [&](size_t bytes) { char* p = w + off; off += (bytes + 255) & ~(size_t)255; return (void*)p; };

    unsigned short* permB_qkv  = (unsigned short*)take(HH * 2);
    unsigned short* permBT_qkv = (unsigned short*)take(HH * 2);
    unsigned short* permB_o    = (unsigned short*)take(HH * 2);
    unsigned short* permBT_o   = (unsigned short*)take(HH * 2);
    unsigned short* xbf        = (unsigned short*)take(BSH * 2);
    float*          colsumf    = (float*)take(H * 4);
    float*          rowsumf    = (float*)take(H * 4);
    unsigned short* metricbf   = (unsigned short*)take(HH * 2);
    float*          Wmf        = (float*)take(HH * 4);
    unsigned short* hardbf     = (unsigned short*)take(HH * 2);
    float*          maskPf     = (float*)take(HH * 4);
    unsigned short* Wmaskbf    = (unsigned short*)take(HH * 2);
    float*          projf      = (float*)take(BSH * 4);
    unsigned short* qbf        = (unsigned short*)take(BSH * 2);
    unsigned short* kbf        = (unsigned short*)take(BSH * 2);
    unsigned short* vtbf       = (unsigned short*)take(BSH * 2);
    unsigned short* attnbf     = (unsigned short*)take(BSH * 2);
    (void)ws_size; (void)n_in; (void)in_sizes; (void)out_size;

    auto blk = [](long n) { return (unsigned)((n + 255) / 256); };

    // --- one-time conversions ---
    k_f32_to_bf16   <<<blk(HH),  256, 0, stream>>>(pqkv, permB_qkv, HH);
    k_transpose_bf16<<<blk(HH),  256, 0, stream>>>(pqkv, permBT_qkv, H);
    k_f32_to_bf16   <<<blk(HH),  256, 0, stream>>>(po,   permB_o, HH);
    k_transpose_bf16<<<blk(HH),  256, 0, stream>>>(po,   permBT_o, H);
    k_f32_to_bf16   <<<blk(BSH), 256, 0, stream>>>(hs,   xbf, BSH);

    dim3 gW(H / 64, H / 256);       // 2048x2048 GEMM grid
    dim3 gP(H / 64, BS / 256);      // 4096x2048 projection grid

    // RIA sparse-mask pipeline: W -> Wmaskbf (bf16 masked weights, NxK row-major)
    auto mask_pipeline = [&](const float* W, const float* sr,
                             const unsigned short* pB, const unsigned short* pBT) {
        k_colsum<<<blk(H), 256, 0, stream>>>(W, colsumf, H, H);
        k_rowsum<<<dim3(H), 256, 0, stream>>>(W, rowsumf, H);
        k_metric<<<blk(HH), 256, 0, stream>>>(W, colsumf, rowsumf, sr, metricbf, H, HH);
        // Wm = metric @ perm   (C = A * B^T with B = perm^T)
        k_gemm_nt<<<gW, 256, 0, stream>>>(metricbf, pBT, Wmf, nullptr, H, H, H);
        k_hard<<<blk(HH / 4), 256, 0, stream>>>(Wmf, hardbf, HH / 4);
        // maskP = hard @ perm^T  (C = A * B^T with B = perm)
        k_gemm_nt<<<gW, 256, 0, stream>>>(hardbf, pB, maskPf, nullptr, H, H, H);
        k_applymask<<<blk(HH), 256, 0, stream>>>(W, maskPf, Wmaskbf, HH);
    };

    // --- Q ---
    mask_pipeline(Wq, sq, permB_qkv, permBT_qkv);
    k_gemm_nt<<<gP, 256, 0, stream>>>(xbf, Wmaskbf, projf, bq, BS, H, H);
    k_rope<<<blk(BSH), 256, 0, stream>>>(projf, pos, qbf, SEQ);

    // --- K ---
    mask_pipeline(Wk, sk, permB_qkv, permBT_qkv);
    k_gemm_nt<<<gP, 256, 0, stream>>>(xbf, Wmaskbf, projf, bk, BS, H, H);
    k_rope<<<blk(BSH), 256, 0, stream>>>(projf, pos, kbf, SEQ);

    // --- V ---
    mask_pipeline(Wv, sv, permB_qkv, permBT_qkv);
    k_gemm_nt<<<gP, 256, 0, stream>>>(xbf, Wmaskbf, projf, bv, BS, H, H);
    k_vt<<<blk(BSH), 256, 0, stream>>>(projf, vtbf, SEQ);

    // --- attention ---
    k_flash<<<dim3(SEQ / 64, BATCH * NHEADS), 128, 0, stream>>>(qbf, kbf, vtbf, attnbf, SEQ);

    // --- O projection ---
    mask_pipeline(Wo, so, permB_o, permBT_o);
    k_gemm_nt<<<gP, 256, 0, stream>>>(attnbf, Wmaskbf, (float*)d_out, nullptr, BS, H, H);
}
